// GATSoftPromptMoE_28948079575201
// MI455X (gfx1250) — matrix operand order repaired and checked by reading, not compiled, not run on hardware
//
#include <hip/hip_runtime.h>
#include <hip/hip_bf16.h>

// ---------------------------------------------------------------------------
// GATSoftPromptMoE for MI455X (gfx1250, wave32).
//  - Dense GEMMs: v_wmma_f32_16x16x32_bf16, operands pre-converted to bf16 in
//    global memory, tiles staged to LDS with global_load_async_to_lds_b128
//    (ASYNCcnt path), B kept K-contiguous per output column -> ds_load_b128.
//  - Edge softmax/scatter: fp32 atomics (memory-bound path).
// ---------------------------------------------------------------------------

#define NN      16384      // nodes
#define DIMF    128        // feature dim (= K of every GEMM)
#define NE      262144     // edges (before self loops)
#define ETOT    (NE + NN)  // edges incl. self loops
#define NEXP    4
#define NHEADS  4

#define APITCH  136        // bf16 elements per LDS row (272B = 17*16B, conflict-free)
#define BPITCH  136

typedef __attribute__((ext_vector_type(16))) __bf16 v16bf;
typedef __attribute__((ext_vector_type(8)))  __bf16 v8bf;
typedef __attribute__((ext_vector_type(4)))  __bf16 v4bf;
typedef __attribute__((ext_vector_type(8)))  float  v8f;

// ---------------------------------------------------------------------------
// fp32 -> bf16 conversion (flat)
// ---------------------------------------------------------------------------
__global__ __launch_bounds__(256)
void cvt_bf16_kernel(const float* __restrict__ src, __bf16* __restrict__ dst, int n) {
    int i = (blockIdx.x * 256 + threadIdx.x) * 4;
    if (i >= n) return;
    const float4 f = *(const float4*)(src + i);
    v4bf o;
    o[0] = (__bf16)f.x; o[1] = (__bf16)f.y; o[2] = (__bf16)f.z; o[3] = (__bf16)f.w;
    *(v4bf*)(dst + i) = o;
}

// fp32 [128, Ncols] -> bf16 transposed [Ncols, 128]
__global__ __launch_bounds__(256)
void cvt_bf16_T_kernel(const float* __restrict__ W, __bf16* __restrict__ WT,
                       int Ncols, int total) {
    int i = blockIdx.x * 256 + threadIdx.x;
    if (i >= total) return;
    int k = i / Ncols, c = i - k * Ncols;
    WT[c * DIMF + k] = (__bf16)W[i];
}

// ---------------------------------------------------------------------------
// WMMA GEMM:  C[M, Ncols] = act(A[M,128] @ W[128,Ncols] (+bias))
// A: bf16 row-major [M,128].  WT: bf16 [Ncols,128] (K contiguous per col).
// Block 256 threads = 8 waves, tile 32(M) x 64(N), K=128 in 4 chunks of 32.
// Tiles staged via global_load_async_to_lds_b128 (ASYNCcnt).
// ---------------------------------------------------------------------------
__global__ __launch_bounds__(256)
void wmma_gemm_k128(const __bf16* __restrict__ Abf, const __bf16* __restrict__ WT,
                    const float* __restrict__ bias, float* __restrict__ C,
                    int Ncols, int do_relu) {
    __shared__ __align__(16) __bf16 Alds[32][APITCH];
    __shared__ __align__(16) __bf16 Blds[64][BPITCH];

    const int tid     = threadIdx.x;
    const int rowBase = blockIdx.y * 32;
    const int colBase = blockIdx.x * 64;

    // LDS byte offsets (generic LDS pointers keep the LDS offset in addr[31:0])
    const unsigned ldsA = (unsigned)(size_t)(&Alds[0][0]);
    const unsigned ldsB = (unsigned)(size_t)(&Blds[0][0]);

    // A tile: 32 rows x 256B  -> 512 x 16B chunks, 2 per thread
#pragma unroll
    for (int c = tid; c < 512; c += 256) {
        int r = c >> 4, o = (c & 15) << 4;
        unsigned la = ldsA + r * (APITCH * 2) + o;
        unsigned go = (unsigned)((rowBase + r) * 256 + o);
        asm volatile("global_load_async_to_lds_b128 %0, %1, %2"
                     :: "v"(la), "v"(go), "s"(Abf) : "memory");
    }
    // B tile: 64 cols x 256B (K-contiguous) -> 1024 x 16B chunks, 4 per thread
#pragma unroll
    for (int c = tid; c < 1024; c += 256) {
        int r = c >> 4, o = (c & 15) << 4;
        unsigned la = ldsB + r * (BPITCH * 2) + o;
        unsigned go = (unsigned)((colBase + r) * 256 + o);
        asm volatile("global_load_async_to_lds_b128 %0, %1, %2"
                     :: "v"(la), "v"(go), "s"(WT) : "memory");
    }
    asm volatile("s_wait_asynccnt 0x0" ::: "memory");
    __syncthreads();

    const int wave  = tid >> 5;
    const int lane  = tid & 31;
    const int half  = lane >> 4;          // 0: lanes 0-15, 1: lanes 16-31
    const int l15   = lane & 15;
    const int waveM = (wave & 1) * 16;    // 2 tiles along M
    const int waveN = (wave >> 1) * 16;   // 4 tiles along N

    v8f acc = {};
#pragma unroll
    for (int kk = 0; kk < 4; ++kk) {
        const int kb = kk * 32;
        // A 16x32 frag: half0 -> K 0..7 / 16..23, half1 -> K 8..15 / 24..31
        const __bf16* ap = &Alds[waveM + l15][kb + half * 8];
        v8bf a0 = *(const v8bf*)ap;
        v8bf a1 = *(const v8bf*)(ap + 16);
        v16bf a = __builtin_shufflevector(a0, a1, 0,1,2,3,4,5,6,7,8,9,10,11,12,13,14,15);
        // B 32x16 frag: half0 -> K 0..7 / 16..23, half1 -> K 8..15 / 24..31
        const __bf16* bp = &Blds[waveN + l15][kb + half * 8];
        v8bf b0 = *(const v8bf*)bp;
        v8bf b1 = *(const v8bf*)(bp + 16);
        v16bf b = __builtin_shufflevector(b0, b1, 0,1,2,3,4,5,6,7,8,9,10,11,12,13,14,15);
        acc = __builtin_amdgcn_wmma_f32_16x16x32_bf16(
            false, a, false, b, (short)0, acc, false, false);
    }

    const bool has_bias = (bias != nullptr);
    const int  col = colBase + waveN + l15;
#pragma unroll
    for (int r = 0; r < 8; ++r) {
        int row = rowBase + waveM + r + half * 8;   // C layout: VGPR r -> M=r (+8 hi half)
        float v = acc[r];
        if (has_bias) v += bias[col];
        if (do_relu)  v = fmaxf(v, 0.0f);
        C[(size_t)row * Ncols + col] = v;
    }
}

// ---------------------------------------------------------------------------
// Gate MLP + top-2 softmax -> dense gates[N,4]
// ---------------------------------------------------------------------------
__global__ __launch_bounds__(128)
void gate_kernel(const float* __restrict__ x,
                 const float* __restrict__ w1, const float* __restrict__ b1,
                 const float* __restrict__ w2, const float* __restrict__ b2,
                 float* __restrict__ gates) {
    const int n = blockIdx.x, j = threadIdx.x;
    __shared__ float xs[DIMF], hs[DIMF], lg[NEXP];
    xs[j] = x[(size_t)n * DIMF + j];
    __syncthreads();
    float acc = b1[j];
    for (int k = 0; k < DIMF; ++k) acc += xs[k] * w1[k * DIMF + j];
    hs[j] = fmaxf(acc, 0.0f);
    __syncthreads();
    if (j < NEXP) {
        float a = b2[j];
        for (int k = 0; k < DIMF; ++k) a += hs[k] * w2[k * NEXP + j];
        lg[j] = a;
    }
    __syncthreads();
    if (j == 0) {
        int i0 = 0;
        for (int t = 1; t < NEXP; ++t) if (lg[t] > lg[i0]) i0 = t;
        int i1 = -1;
        for (int t = 0; t < NEXP; ++t) {
            if (t == i0) continue;
            if (i1 < 0 || lg[t] > lg[i1]) i1 = t;
        }
        float m  = fmaxf(lg[i0], lg[i1]);
        float e0 = __expf(lg[i0] - m), e1 = __expf(lg[i1] - m);
        float s  = e0 + e1;
        float g[NEXP] = {0.f, 0.f, 0.f, 0.f};
        g[i0] = e0 / s; g[i1] = e1 / s;
        for (int t = 0; t < NEXP; ++t) gates[n * NEXP + t] = g[t];
    }
}

// ---------------------------------------------------------------------------
// SoftPrompt fusion + LayerNorm (per node, 128 threads)
// ---------------------------------------------------------------------------
__global__ __launch_bounds__(128)
void fuse_ln_kernel(const float* __restrict__ spec, const float* __restrict__ prompt,
                    const float* __restrict__ fusion, const float* __restrict__ g,
                    const float* __restrict__ b, float* __restrict__ xp) {
    const int n = blockIdx.x, j = threadIdx.x;
    __shared__ float red[4];
    float pm = 0.f;
#pragma unroll
    for (int p = 0; p < 8; ++p) pm += prompt[p * DIMF + j];
    pm *= 0.125f;
    const float fw = 1.0f / (1.0f + __expf(-fusion[0]));
    const float t  = (1.0f - fw) * spec[(size_t)n * DIMF + j] + fw * pm;

    float s = t;
    for (int o = 16; o; o >>= 1) s += __shfl_xor(s, o, 32);
    if ((j & 31) == 0) red[j >> 5] = s;
    __syncthreads();
    const float mu = (red[0] + red[1] + red[2] + red[3]) * (1.0f / DIMF);
    const float d  = t - mu;
    float s2 = d * d;
    for (int o = 16; o; o >>= 1) s2 += __shfl_xor(s2, o, 32);
    __syncthreads();
    if ((j & 31) == 0) red[j >> 5] = s2;
    __syncthreads();
    const float var = (red[0] + red[1] + red[2] + red[3]) * (1.0f / DIMF);
    xp[(size_t)n * DIMF + j] = d * rsqrtf(var + 1e-5f) * g[j] + b[j];
}

// ---------------------------------------------------------------------------
// Attention coefficients (1 wave per n,h)
// ---------------------------------------------------------------------------
__global__ __launch_bounds__(256)
void att_kernel(const float* __restrict__ xh, const float* __restrict__ as_w,
                const float* __restrict__ ad_w, int H, int nWaves,
                float* __restrict__ a_s, float* __restrict__ a_d) {
    const int wave = (blockIdx.x * blockDim.x + threadIdx.x) >> 5;
    const int lane = threadIdx.x & 31;
    if (wave >= nWaves) return;
    const int n = wave / H, h = wave % H;
    const float* row = xh + ((size_t)n * H + h) * DIMF;
    float s = 0.f, d = 0.f;
    for (int c = lane; c < DIMF; c += 32) {
        float v = row[c];
        s += v * as_w[h * DIMF + c];
        d += v * ad_w[h * DIMF + c];
    }
    for (int o = 16; o; o >>= 1) { s += __shfl_xor(s, o, 32); d += __shfl_xor(d, o, 32); }
    if (lane == 0) { a_s[n * H + h] = s; a_d[n * H + h] = d; }
}

// ---------------------------------------------------------------------------
// Edge passes: ordered-int float max, exp+sum, weighted scatter
// ---------------------------------------------------------------------------
__device__ __forceinline__ unsigned fkey(float f) {
    unsigned b = __float_as_uint(f);
    return (b & 0x80000000u) ? ~b : (b | 0x80000000u);
}
__device__ __forceinline__ float fkey_dec(unsigned k) {
    unsigned b = (k & 0x80000000u) ? (k & 0x7FFFFFFFu) : ~k;
    return __uint_as_float(b);
}
__device__ __forceinline__ void edge_pair(const int* ei, int idx, int& src, int& dst) {
    if (idx < NE) { src = ei[idx]; dst = ei[NE + idx]; }
    else          { src = dst = idx - NE; }
}

__global__ __launch_bounds__(256)
void edge_max_kernel(const int* __restrict__ ei, int H,
                     const float* __restrict__ a_s, const float* __restrict__ a_d,
                     unsigned* __restrict__ mkey) {
    const int idx = blockIdx.x * blockDim.x + threadIdx.x;
    if (idx >= ETOT) return;
    int src, dst; edge_pair(ei, idx, src, dst);
    for (int h = 0; h < H; ++h) {
        float e = a_s[src * H + h] + a_d[dst * H + h];
        e = (e < 0.f) ? 0.2f * e : e;                 // leaky_relu 0.2
        atomicMax(&mkey[dst * H + h], fkey(e));
    }
}

__global__ __launch_bounds__(256)
void edge_exp_kernel(const int* __restrict__ ei, int H,
                     const float* __restrict__ a_s, const float* __restrict__ a_d,
                     const unsigned* __restrict__ mkey,
                     float* __restrict__ exbuf, float* __restrict__ denom) {
    const int idx = blockIdx.x * blockDim.x + threadIdx.x;
    if (idx >= ETOT) return;
    int src, dst; edge_pair(ei, idx, src, dst);
    for (int h = 0; h < H; ++h) {
        float e = a_s[src * H + h] + a_d[dst * H + h];
        e = (e < 0.f) ? 0.2f * e : e;
        float ex = __expf(e - fkey_dec(mkey[dst * H + h]));
        exbuf[(size_t)idx * H + h] = ex;
        atomicAdd(&denom[dst * H + h], ex);
    }
}

__global__ __launch_bounds__(256)
void edge_scatter_kernel(const int* __restrict__ ei, int H,
                         const float* __restrict__ exbuf, const float* __restrict__ denom,
                         const float* __restrict__ xh, float* __restrict__ accum) {
    const int wave = (blockIdx.x * blockDim.x + threadIdx.x) >> 5;
    const int lane = threadIdx.x & 31;
    if (wave >= ETOT) return;
    int src, dst; edge_pair(ei, wave, src, dst);
    const int HC = H * DIMF;
    __builtin_prefetch(&xh[(size_t)src * HC], 0, 3);
    for (int t = lane; t < HC; t += 32) {
        int h = t >> 7;
        float alpha = exbuf[(size_t)wave * H + h] / (denom[dst * H + h] + 1e-16f);
        atomicAdd(&accum[(size_t)dst * HC + t], alpha * xh[(size_t)src * HC + t]);
    }
}

// ---------------------------------------------------------------------------
// Head-mean + bias (+residual) + LayerNorm + relu
// ---------------------------------------------------------------------------
__global__ __launch_bounds__(128)
void head_ln_kernel(const float* __restrict__ accum, int H,
                    const float* __restrict__ bias, const float* __restrict__ residual,
                    const float* __restrict__ g, const float* __restrict__ b,
                    float* __restrict__ out) {
    const int n = blockIdx.x, j = threadIdx.x;
    __shared__ float red[4];
    float t = 0.f;
    for (int h = 0; h < H; ++h) t += accum[((size_t)n * H + h) * DIMF + j];
    t = t * (1.0f / (float)H) + bias[j];
    if (residual) t += residual[(size_t)n * DIMF + j];

    float s = t;
    for (int o = 16; o; o >>= 1) s += __shfl_xor(s, o, 32);
    if ((j & 31) == 0) red[j >> 5] = s;
    __syncthreads();
    const float mu = (red[0] + red[1] + red[2] + red[3]) * (1.0f / DIMF);
    const float d  = t - mu;
    float s2 = d * d;
    for (int o = 16; o; o >>= 1) s2 += __shfl_xor(s2, o, 32);
    __syncthreads();
    if ((j & 31) == 0) red[j >> 5] = s2;
    __syncthreads();
    const float var = (red[0] + red[1] + red[2] + red[3]) * (1.0f / DIMF);
    out[(size_t)n * DIMF + j] = fmaxf(d * rsqrtf(var + 1e-5f) * g[j] + b[j], 0.0f);
}

// Final: y (+)= gates[:,e] * (accum + gat3_b)   (H=1, no LN/relu)
__global__ __launch_bounds__(128)
void final_accum_kernel(const float* __restrict__ accum, const float* __restrict__ bias,
                        const float* __restrict__ gates, int e, int first,
                        float* __restrict__ y) {
    const int n = blockIdx.x, j = threadIdx.x;
    float v = gates[n * NEXP + e] * (accum[(size_t)n * DIMF + j] + bias[j]);
    if (first) y[(size_t)n * DIMF + j] = v;
    else       y[(size_t)n * DIMF + j] += v;
}

__global__ __launch_bounds__(256)
void fill_kernel(float* __restrict__ p, float v, int n) {
    int i = blockIdx.x * blockDim.x + threadIdx.x;
    if (i < n) p[i] = v;
}

// ---------------------------------------------------------------------------
// Host orchestration
// ---------------------------------------------------------------------------
extern "C" void kernel_launch(void* const* d_in, const int* in_sizes, int n_in,
                              void* d_out, int out_size, void* d_ws, size_t ws_size,
                              hipStream_t stream) {
    // Inputs in setup_inputs() dict order (nested dicts flattened in insertion order)
    const float* x        = (const float*)d_in[0];
    const int*   ei       = (const int*)  d_in[1];
    const float* gate_w1  = (const float*)d_in[2];
    const float* gate_b1  = (const float*)d_in[3];
    const float* gate_w2  = (const float*)d_in[4];
    const float* gate_b2  = (const float*)d_in[5];
    const float* prompt   = (const float*)d_in[6];   // [4,8,128]
    const float* sp_w1    = (const float*)d_in[7];   // [4,128,128]
    const float* sp_b1    = (const float*)d_in[8];
    const float* sp_w2    = (const float*)d_in[9];
    const float* sp_b2    = (const float*)d_in[10];
    const float* fusion   = (const float*)d_in[11];  // [4]
    const float* ln_p_g   = (const float*)d_in[12];
    const float* ln_p_b   = (const float*)d_in[13];
    const float* gat1_W   = (const float*)d_in[14];  // [4,128,512]
    const float* gat1_as  = (const float*)d_in[15];  // [4,4,128]
    const float* gat1_ad  = (const float*)d_in[16];
    const float* gat1_b   = (const float*)d_in[17];
    const float* gat2_W   = (const float*)d_in[18];
    const float* gat2_as  = (const float*)d_in[19];
    const float* gat2_ad  = (const float*)d_in[20];
    const float* gat2_b   = (const float*)d_in[21];
    const float* gat3_W   = (const float*)d_in[22];  // [4,128,128]
    const float* gat3_as  = (const float*)d_in[23];  // [4,1,128]
    const float* gat3_ad  = (const float*)d_in[24];
    const float* gat3_b   = (const float*)d_in[25];
    const float* ln1_g    = (const float*)d_in[26];
    const float* ln1_b    = (const float*)d_in[27];
    const float* ln2_g    = (const float*)d_in[28];
    const float* ln2_b    = (const float*)d_in[29];
    float* y = (float*)d_out;
    (void)ws_size; (void)n_in; (void)in_sizes; (void)out_size;

    // Workspace carve (bytes, 256B aligned)
    char* wsb = (char*)d_ws;
    size_t off = 0;
    auto carve = [&](size_t bytes) {
        void* p = wsb + off;
        off += (bytes + 255) & ~(size_t)255;
        return p;
    };
    float*    gates = (float*)   carve((size_t)NN * NEXP * 4);
    float*    xp    = (float*)   carve((size_t)NN * DIMF * 4);
    float*    h1    = (float*)   carve((size_t)NN * DIMF * 4);
    float*    h2    = (float*)   carve((size_t)NN * DIMF * 4);
    float*    tmpA  = (float*)   carve((size_t)NN * DIMF * 4);
    float*    tmpB  = (float*)   carve((size_t)NN * DIMF * 4);
    float*    xh    = (float*)   carve((size_t)NN * NHEADS * DIMF * 4);
    float*    accum = (float*)   carve((size_t)NN * NHEADS * DIMF * 4);
    float*    a_s   = (float*)   carve((size_t)NN * NHEADS * 4);
    float*    a_d   = (float*)   carve((size_t)NN * NHEADS * 4);
    unsigned* mkey  = (unsigned*)carve((size_t)NN * NHEADS * 4);
    float*    denom = (float*)   carve((size_t)NN * NHEADS * 4);
    float*    exbuf = (float*)   carve((size_t)ETOT * NHEADS * 4);
    __bf16*   xbf   = (__bf16*)  carve((size_t)NN * DIMF * 2);  // bf16 copy of x
    __bf16*   Abf   = (__bf16*)  carve((size_t)NN * DIMF * 2);  // bf16 copy of current A
    __bf16*   WbfT  = (__bf16*)  carve((size_t)DIMF * NHEADS * DIMF * 2); // [Ncols,128]

    auto cvtA = [&](const float* src, __bf16* dst) {
        const int n = NN * DIMF;
        cvt_bf16_kernel<<<(n / 4 + 255) / 256, 256, 0, stream>>>(src, dst, n);
    };
    auto cvtW = [&](const float* W, int Ncols) {
        const int total = DIMF * Ncols;
        cvt_bf16_T_kernel<<<(total + 255) / 256, 256, 0, stream>>>(W, WbfT, Ncols, total);
    };
    auto gemm = [&](const __bf16* Ab, const float* bias, float* C, int Ncols, int relu) {
        dim3 grid(Ncols / 64, NN / 32);
        wmma_gemm_k128<<<grid, 256, 0, stream>>>(Ab, WbfT, bias, C, Ncols, relu);
    };
    auto fill = [&](float* p, float v, size_t n) {
        fill_kernel<<<(unsigned)((n + 255) / 256), 256, 0, stream>>>(p, v, (int)n);
    };

    // Gating + bf16 copy of x (reused by all 4 experts' first GEMM)
    gate_kernel<<<NN, 128, 0, stream>>>(x, gate_w1, gate_b1, gate_w2, gate_b2, gates);
    cvtA(x, xbf);

    for (int e = 0; e < NEXP; ++e) {
        const size_t wo = (size_t)e * DIMF * DIMF;           // 128x128 weight stride
        const size_t wg = (size_t)e * DIMF * NHEADS * DIMF;  // 128x512 weight stride
        const size_t bo = (size_t)e * DIMF;

        // SoftPrompt MLP + fusion LN
        cvtW(sp_w1 + wo, DIMF);
        gemm(xbf, sp_b1 + bo, tmpA, DIMF, 1);
        cvtA(tmpA, Abf);
        cvtW(sp_w2 + wo, DIMF);
        gemm(Abf, sp_b2 + bo, tmpB, DIMF, 0);
        fuse_ln_kernel<<<NN, 128, 0, stream>>>(tmpB, prompt + (size_t)e * 8 * DIMF,
                                               fusion + e, ln_p_g + bo, ln_p_b + bo, xp);

        auto gat = [&](const float* hin, const float* W, const float* as_w,
                       const float* ad_w, int H) {
            const int HC = H * DIMF;
            cvtA(hin, Abf);
            cvtW(W, HC);
            gemm(Abf, nullptr, xh, HC, 0);
            const int nWaves = NN * H;
            att_kernel<<<(nWaves * 32) / 256, 256, 0, stream>>>(xh, as_w, ad_w, H, nWaves, a_s, a_d);
            fill((float*)mkey, 0.0f, (size_t)NN * H);   // key 0 < key(any real float)
            fill(denom,        0.0f, (size_t)NN * H);
            fill(accum,        0.0f, (size_t)NN * HC);
            edge_max_kernel<<<(ETOT + 255) / 256, 256, 0, stream>>>(ei, H, a_s, a_d, mkey);
            edge_exp_kernel<<<(ETOT + 255) / 256, 256, 0, stream>>>(ei, H, a_s, a_d, mkey, exbuf, denom);
            edge_scatter_kernel<<<(ETOT * 32) / 256, 256, 0, stream>>>(ei, H, exbuf, denom, xh, accum);
        };

        // GAT layer 1
        gat(xp, gat1_W + wg, gat1_as + (size_t)e * NHEADS * DIMF,
            gat1_ad + (size_t)e * NHEADS * DIMF, NHEADS);
        head_ln_kernel<<<NN, 128, 0, stream>>>(accum, NHEADS, gat1_b + bo, nullptr,
                                               ln1_g + bo, ln1_b + bo, h1);
        // GAT layer 2 (residual h1)
        gat(h1, gat2_W + wg, gat2_as + (size_t)e * NHEADS * DIMF,
            gat2_ad + (size_t)e * NHEADS * DIMF, NHEADS);
        head_ln_kernel<<<NN, 128, 0, stream>>>(accum, NHEADS, gat2_b + bo, h1,
                                               ln2_g + bo, ln2_b + bo, h2);
        // GAT layer 3 (H=1), then gated accumulation into y
        gat(h2, gat3_W + wo, gat3_as + bo, gat3_ad + bo, 1);
        final_accum_kernel<<<NN, 128, 0, stream>>>(accum, gat3_b + bo, gates, e,
                                                   (e == 0) ? 1 : 0, y);
    }
}
